// MultiHeadAttention_17128329577097
// MI455X (gfx1250) — compile-verified, matrix operand order checked by reading, not run on hardware
//
#include <hip/hip_runtime.h>
#include <hip/hip_bf16.h>
#include <cstdint>
#include <cstddef>

#define HIDDEN 768
#define HEADS 12
#define DHEAD 64
#define BATCH 8
#define SEQ 1024
#define LN_EPS 1e-5f

typedef __bf16 bf16_t;
typedef __attribute__((ext_vector_type(8)))  __bf16 v8bf;
typedef __attribute__((ext_vector_type(16))) __bf16 v16bf;
typedef __attribute__((ext_vector_type(8)))  float  v8f;
typedef __attribute__((ext_vector_type(4)))  unsigned int v4u;
typedef __attribute__((ext_vector_type(4)))  int v4i;
typedef __attribute__((ext_vector_type(8)))  int v8i;

// TDM availability: device pass only; host pass uses the fallback copy.
#if defined(__HIP_DEVICE_COMPILE__) && __has_builtin(__builtin_amdgcn_tensor_load_to_lds) && \
    __has_builtin(__builtin_amdgcn_s_wait_tensorcnt)
#define HAS_TDM 1
#else
#define HAS_TDM 0
#endif

// ---------------------------------------------------------------------------
// Fragment loaders for V_WMMA_F32_16X16X32_BF16 (wave32), ISA 7.12.2 layout:
// lane L -> row/col = L&15, half = L>>4;
//   slots 0..7  hold K = kb + 8*half + i
//   slots 8..15 hold K = kb + 16 + 8*half + i
// Loading *rows* of two row-major matrices with the same mapping computes
// A * B^T (the x@W.T / Q@K.T / P@Vt patterns below).
// ---------------------------------------------------------------------------
__device__ __forceinline__ v16bf frag_from_bf16(const bf16_t* base, int stride, int kb, int lane) {
  const int r = lane & 15, h = lane >> 4;
  const bf16_t* p = base + (size_t)r * stride + kb + 8 * h;
  v8bf lo = *(const v8bf*)(p);
  v8bf hi = *(const v8bf*)(p + 16);
  v16bf out;
#pragma unroll
  for (int i = 0; i < 8; ++i) { out[i] = lo[i]; out[i + 8] = hi[i]; }
  return out;
}

__device__ __forceinline__ v16bf frag_from_f32(const float* base, int stride, int kb, int lane) {
  const int r = lane & 15, h = lane >> 4;
  const float* p = base + (size_t)r * stride + kb + 8 * h;
  v16bf out;
#pragma unroll
  for (int i = 0; i < 8; ++i) { out[i] = (bf16_t)p[i]; out[i + 8] = (bf16_t)p[16 + i]; }
  return out;
}

__device__ __forceinline__ v8f wmma_bf16(v16bf a, v16bf b, v8f c) {
  // (neg_a, A, neg_b, B, c_mod, C, reuse_a, reuse_b)
  return __builtin_amdgcn_wmma_f32_16x16x32_bf16(false, a, false, b, (short)0, c, false, false);
}

// ---------------------------------------------------------------------------
// Kernel 1: X @ W.T + bias for one of Q/K/V.
// grid = (512 row-tiles, 3 head-groups), block = 128 (4 waves).
// Each wave owns one full head (64 cols = 4 N-tiles, 4 accumulators) so the
// A fragment is loaded once per k-step and reused for 4 WMMAs.
// V output is stored transposed per head: Vt[bh][d][sk].
// ---------------------------------------------------------------------------
__global__ void proj_kernel(const float* __restrict__ X, const float* __restrict__ W,
                            const float* __restrict__ bias, bf16_t* __restrict__ dst,
                            int vtranspose) {
  const int lane = threadIdx.x & 31;
  const int wave = threadIdx.x >> 5;
  const int rowbase = blockIdx.x * 16;
  const int head = blockIdx.y * 4 + wave;        // 0..11
  const int colbase = head * DHEAD;
  const int nl = lane & 15, h = lane >> 4;

  const float* Ab = X + (size_t)rowbase * HIDDEN;
  const float* Bb = W + (size_t)colbase * HIDDEN;

  v8f acc[4] = {};
  for (int k = 0; k < HIDDEN; k += 32) {
    v16bf a = frag_from_f32(Ab, HIDDEN, k, lane);
#pragma unroll
    for (int t = 0; t < 4; ++t) {
      v16bf b = frag_from_f32(Bb + (size_t)t * 16 * HIDDEN, HIDDEN, k, lane);
      acc[t] = wmma_bf16(a, b, acc[t]);
    }
  }

  if (!vtranspose) {
    // Qh/Kh layout: [b][h][s][d] bf16
#pragma unroll
    for (int t = 0; t < 4; ++t) {
      const float bia = bias[colbase + t * 16 + nl];
#pragma unroll
      for (int r = 0; r < 8; ++r) {
        int m = rowbase + r + 8 * h;
        int bidx = m >> 10, sq = m & (SEQ - 1);
        dst[(((size_t)bidx * HEADS + head) * SEQ + sq) * DHEAD + t * 16 + nl] =
            (bf16_t)(acc[t][r] + bia);
      }
    }
  } else {
    // Vt layout: [b][h][d][sk] bf16 — 8 consecutive sk per lane -> 16B store
    int m0 = rowbase + 8 * h;
    int bidx = m0 >> 10, sk0 = m0 & (SEQ - 1);
#pragma unroll
    for (int t = 0; t < 4; ++t) {
      const float bia = bias[colbase + t * 16 + nl];
      int d = t * 16 + nl;
      v8bf pk;
#pragma unroll
      for (int r = 0; r < 8; ++r) pk[r] = (bf16_t)(acc[t][r] + bia);
      *(v8bf*)(dst + (((size_t)bidx * HEADS + head) * DHEAD + d) * SEQ + sk0) = pk;
    }
  }
}

// ---------------------------------------------------------------------------
// Kernel 2: attention for one (b, h, 16-query tile). 4 waves.
//   Stage:   Q tile (16x64 bf16) -> LDS via TENSOR_LOAD_TO_LDS (TDM) + wait
//   Phase A: S^T = K·Q^T (M=sk, N=q) -> f32 LDS score[16][1024] (pre-scaled);
//            Q fragments are loop-invariant and hoisted.
//   Phase B: masked softmax per q row (8 threads/row) -> P[16][1024] bf16
//            (unnormalized); Vt panel prefetched meanwhile.
//   Phase C: ctx = P · V (A = P rows, B = Vt rows), scaled by 1/rowsum.
// ---------------------------------------------------------------------------
__global__ void attn_kernel(const bf16_t* __restrict__ Qh, const bf16_t* __restrict__ Kh,
                            const bf16_t* __restrict__ Vt, const unsigned char* __restrict__ mask,
                            bf16_t* __restrict__ ctx) {
  __shared__ float  score[16][SEQ];   // 64 KB
  __shared__ bf16_t P[16][SEQ];       // 32 KB
  __shared__ bf16_t Qs[16][DHEAD];    // 2 KB, TDM-staged
  __shared__ float  redmax[16][8];
  __shared__ float  redsum[16][8];
  __shared__ float  rowmax[16];
  __shared__ float  rowrcp[16];

  const int tid = threadIdx.x;
  const int lane = tid & 31, wave = tid >> 5;
  const int bh = blockIdx.y;                  // b*HEADS + h
  const int b = bh / HEADS, head = bh % HEADS;
  const int qbase = blockIdx.x * 16;
  const int nl = lane & 15, h = lane >> 4;

  const bf16_t* Qsrc = Qh + ((size_t)bh * SEQ + qbase) * DHEAD;

#if HAS_TDM
  if (wave == 0) {
    // Tensor DMA descriptor (ISA 8.3/8.4): 2D tile, 2-byte elements,
    // tile 64 x 16, tensor row stride 64, dest = Qs in LDS.
    const uint64_t ga = (uint64_t)(uintptr_t)Qsrc;
    const unsigned int lds_off = (unsigned int)(uintptr_t)(&Qs[0][0]);
    v4u g0;
    g0[0] = 1u;                                   // count = 1 (valid D#)
    g0[1] = lds_off;                              // lds_addr
    g0[2] = (unsigned int)ga;                     // global_addr[31:0]
    g0[3] = (unsigned int)(ga >> 32) | 0x80000000u; // global_addr[56:32] | type=2
    v8i g1;
    g1[0] = 0x00010000;                           // data_size = 1 (2 bytes)
    g1[1] = (int)(64u << 16);                     // tensor_dim0 = 64
    g1[2] = (int)(1024u << 16);                   // tensor_dim1 = 1024
    g1[3] = (int)(64u << 16);                     // tile_dim0 = 64
    g1[4] = 16;                                   // tile_dim1 = 16
    g1[5] = 64;                                   // tensor_dim0_stride = 64
    g1[6] = 0;
    g1[7] = 0;
    v4i z4 = {};
#if __clang_major__ >= 23
    v8i z8 = {};
    __builtin_amdgcn_tensor_load_to_lds(g0, g1, z4, z4, z8, 0);
#else
    __builtin_amdgcn_tensor_load_to_lds(g0, g1, z4, z4, 0);
#endif
    __builtin_amdgcn_s_wait_tensorcnt((short)0);
  }
#else
  for (int i = tid; i < 16 * DHEAD / 2; i += 128)
    ((unsigned int*)&Qs[0][0])[i] = ((const unsigned int*)Qsrc)[i];
#endif
  __syncthreads();

  // Q fragments are invariant across all sk tiles: hoist them.
  const v16bf qa0 = frag_from_bf16(&Qs[0][0], DHEAD, 0, lane);
  const v16bf qa1 = frag_from_bf16(&Qs[0][0], DHEAD, 32, lane);

  // Phase A: each wave covers a 256-wide sk stripe (16 tiles x 2 WMMAs)
  for (int tt = 0; tt < 16; ++tt) {
    int skbase = wave * 256 + tt * 16;
    const bf16_t* Krow = Kh + ((size_t)bh * SEQ + skbase) * DHEAD;
    v8f acc = {};
    acc = wmma_bf16(frag_from_bf16(Krow, DHEAD, 0, lane), qa0, acc);
    acc = wmma_bf16(frag_from_bf16(Krow, DHEAD, 32, lane), qa1, acc);
    int sk0 = skbase + 8 * h;  // C frag: lane -> q = nl, VGPR r -> sk = sk0 + r
#pragma unroll
    for (int r = 0; r < 8; ++r) score[nl][sk0 + r] = acc[r] * 0.125f;  // 1/sqrt(64)
  }

  // Prefetch the Vt panel for this (b,h) while softmax runs:
  // 128 KB = 128 threads x 4 prefetches x 256 B.
  {
    const bf16_t* vb = Vt + (size_t)bh * DHEAD * SEQ + (size_t)tid * 512;
#pragma unroll
    for (int i = 0; i < 4; ++i) __builtin_prefetch(vb + i * 128, 0, 1);
  }
  __syncthreads();

  // Phase B: masked softmax (uniform control flow; barriers hit by all threads)
  {
    const int q = tid >> 3;    // 0..15
    const int part = tid & 7;  // 0..7, each scans 128 sk
    const int s0 = part * 128;
    const unsigned char* mrow = mask + ((size_t)b * SEQ + (qbase + q)) * SEQ;
    float mx = -3.0e38f;
    for (int s = s0; s < s0 + 128; ++s)
      if (!mrow[s]) mx = fmaxf(mx, score[q][s]);
    redmax[q][part] = mx;
    __syncthreads();
    if (part == 0) {
      float m2 = redmax[q][0];
      for (int i = 1; i < 8; ++i) m2 = fmaxf(m2, redmax[q][i]);
      rowmax[q] = m2;
    }
    __syncthreads();
    const float rm = rowmax[q];
    float sum = 0.f;
    for (int s = s0; s < s0 + 128; ++s) {
      float e = mrow[s] ? 0.f : __expf(score[q][s] - rm);
      P[q][s] = (bf16_t)e;
      sum += e;
    }
    redsum[q][part] = sum;
    __syncthreads();
    if (part == 0) {
      float t = 0.f;
      for (int i = 0; i < 8; ++i) t += redsum[q][i];
      rowrcp[q] = 1.0f / t;
    }
    __syncthreads();
  }

  // Phase C: each wave computes one 16-wide d tile; K = full 1024 sk
  {
    const bf16_t* Vrow = Vt + ((size_t)bh * DHEAD + wave * 16) * SEQ;
    v8f acc = {};
    for (int kk = 0; kk < SEQ; kk += 32) {
      v16bf a  = frag_from_bf16(&P[0][0], SEQ, kk, lane);  // rows = q
      v16bf b2 = frag_from_bf16(Vrow, SEQ, kk, lane);      // rows = d (Vt)
      acc = wmma_bf16(a, b2, acc);
    }
    const int col = head * DHEAD + wave * 16 + nl;
#pragma unroll
    for (int r = 0; r < 8; ++r) {
      int q = r + 8 * h;
      float v = acc[r] * rowrcp[q];
      size_t row = (size_t)b * SEQ + qbase + q;
      ctx[row * (HEADS * DHEAD) + col] = (bf16_t)v;
    }
  }
}

// ---------------------------------------------------------------------------
// Kernel 3: out = ctx @ Wo.T + b_o + residual, then LayerNorm over 768.
// One block per 16 rows; 8 waves; k-outer / 6-N-tile-inner so the A fragment
// is loaded once per k-step and reused for 6 WMMAs. LN fused via LDS.
// ---------------------------------------------------------------------------
__global__ void oproj_ln_kernel(const bf16_t* __restrict__ ctx, const float* __restrict__ Wo,
                                const float* __restrict__ bo, const float* __restrict__ residual,
                                const float* __restrict__ ln_g, const float* __restrict__ ln_b,
                                float* __restrict__ out) {
  __shared__ float xs[16][HIDDEN + 16];  // ~50 KB, padded
  __shared__ float redsum[16][16];
  __shared__ float redsq[16][16];
  __shared__ float s_mu[16];
  __shared__ float s_inv[16];

  const int tid = threadIdx.x;
  const int lane = tid & 31, wave = tid >> 5;
  const int rowbase = blockIdx.x * 16;
  const int nl = lane & 15, h = lane >> 4;

  const bf16_t* Ab = ctx + (size_t)rowbase * HIDDEN;
  const float* Bb = Wo + (size_t)(wave * 6 * 16) * HIDDEN;

  v8f acc[6] = {};
  for (int k = 0; k < HIDDEN; k += 32) {
    v16bf a = frag_from_bf16(Ab, HIDDEN, k, lane);
#pragma unroll
    for (int j = 0; j < 6; ++j)
      acc[j] = wmma_bf16(a, frag_from_f32(Bb + (size_t)j * 16 * HIDDEN, HIDDEN, k, lane), acc[j]);
  }

#pragma unroll
  for (int j = 0; j < 6; ++j) {
    const int n = (wave * 6 + j) * 16 + nl;
    const float bia = bo[n];
#pragma unroll
    for (int r = 0; r < 8; ++r) {
      int mr = r + 8 * h;
      xs[mr][n] = acc[j][r] + bia + residual[(size_t)(rowbase + mr) * HIDDEN + n];
    }
  }
  __syncthreads();

  const int row = tid >> 4;  // 16 rows x 16 threads
  const int t = tid & 15;
  float s = 0.f, sq = 0.f;
  for (int c = t; c < HIDDEN; c += 16) { float v = xs[row][c]; s += v; sq += v * v; }
  redsum[row][t] = s;
  redsq[row][t] = sq;
  __syncthreads();
  if (t == 0) {
    float ts = 0.f, tq = 0.f;
    for (int i = 0; i < 16; ++i) { ts += redsum[row][i]; tq += redsq[row][i]; }
    float mu = ts / (float)HIDDEN;
    float var = tq / (float)HIDDEN - mu * mu;
    s_mu[row] = mu;
    s_inv[row] = rsqrtf(var + LN_EPS);
  }
  __syncthreads();
  const float mu = s_mu[row], inv = s_inv[row];
  for (int c = t; c < HIDDEN; c += 16) {
    out[(size_t)(rowbase + row) * HIDDEN + c] =
        (xs[row][c] - mu) * inv * ln_g[c] + ln_b[c];
  }
}

// ---------------------------------------------------------------------------
extern "C" void kernel_launch(void* const* d_in, const int* in_sizes, int n_in,
                              void* d_out, int out_size, void* d_ws, size_t ws_size,
                              hipStream_t stream) {
  const float* query = (const float*)d_in[0];
  const float* keyi  = (const float*)d_in[1];
  const float* value = (const float*)d_in[2];
  const float* w_q   = (const float*)d_in[3];
  const float* b_q   = (const float*)d_in[4];
  const float* w_k   = (const float*)d_in[5];
  const float* b_k   = (const float*)d_in[6];
  const float* w_v   = (const float*)d_in[7];
  const float* b_v   = (const float*)d_in[8];
  const float* w_o   = (const float*)d_in[9];
  const float* b_o   = (const float*)d_in[10];
  const float* ln_g  = (const float*)d_in[11];
  const float* ln_b  = (const float*)d_in[12];
  const unsigned char* mask = (const unsigned char*)d_in[13];
  (void)in_sizes; (void)n_in; (void)out_size; (void)ws_size;

  char* ws = (char*)d_ws;
  size_t off = 0;
  auto take = [&](size_t bytes) -> char* {
    char* p = ws + off;
    off = (off + bytes + 255) & ~(size_t)255;
    return p;
  };
  const size_t qkv_elems = (size_t)BATCH * HEADS * SEQ * DHEAD;        // 6.29M
  bf16_t* Qh  = (bf16_t*)take(qkv_elems * sizeof(bf16_t));             // [b][h][s][d]
  bf16_t* Kh  = (bf16_t*)take(qkv_elems * sizeof(bf16_t));             // [b][h][s][d]
  bf16_t* Vt  = (bf16_t*)take(qkv_elems * sizeof(bf16_t));             // [b][h][d][s]
  bf16_t* ctx = (bf16_t*)take((size_t)BATCH * SEQ * HIDDEN * sizeof(bf16_t));

  dim3 gproj(BATCH * SEQ / 16, 3);
  proj_kernel<<<gproj, 128, 0, stream>>>(query, w_q, b_q, Qh, 0);
  proj_kernel<<<gproj, 128, 0, stream>>>(keyi,  w_k, b_k, Kh, 0);
  proj_kernel<<<gproj, 128, 0, stream>>>(value, w_v, b_v, Vt, 1);

  attn_kernel<<<dim3(SEQ / 16, BATCH * HEADS), 128, 0, stream>>>(Qh, Kh, Vt, mask, ctx);

  oproj_ln_kernel<<<dim3(BATCH * SEQ / 16), 256, 0, stream>>>(ctx, w_o, b_o, query,
                                                              ln_g, ln_b, (float*)d_out);
}
